// GCNSet2SetNoGCNNet_70583492542780
// MI455X (gfx1250) — matrix operand order repaired and checked by reading, not compiled, not run on hardware
//
#include <hip/hip_runtime.h>
#include <hip/hip_bf16.h>
#include <math.h>

#define BG   1024   // number of graphs
#define HDIM 64     // feature / hidden dim
#define QS   128    // q_star width = 2H
#define GW   256    // gate width = 4H
#define STEPS 5

typedef __attribute__((ext_vector_type(16))) _Float16 v16h;
typedef __attribute__((ext_vector_type(8)))  float    v8f;
typedef unsigned long long u64;

// ---------------------------------------------------------------------------
// WMMA tile loaders, per CDNA5 ISA §7.12.2 layouts (wave32).
// A (16x32 f16): lane&15 = M row; lane<16 -> halves[0..7]=K0..7, [8..15]=K16..23
//                lane>=16 -> halves[0..7]=K8..15, [8..15]=K24..31
// B (32x16 f16): lane&15 = N col; same K order as A (lane<16 -> K0..7,K16..23 ; lane>=16 -> K8..15,K24..31)
// C/D (16x16 f32): VGPR v: lane<16 -> M=v, N=lane ; lane>=16 -> M=v+8, N=lane-16
// ---------------------------------------------------------------------------
__device__ inline v16h load_a_f32(const float* __restrict__ A, int lda,
                                  int m0, int k0, int lane) {
  int m  = m0 + (lane & 15);
  int kb = k0 + ((lane & 16) ? 8 : 0);
  const float* row = A + (size_t)m * lda;
  v16h a;
#pragma unroll
  for (int i = 0; i < 8; ++i) a[i]     = (_Float16)row[kb + i];
#pragma unroll
  for (int i = 0; i < 8; ++i) a[8 + i] = (_Float16)row[kb + 16 + i];
  return a;
}

__device__ inline v16h load_a_f16(const _Float16* __restrict__ A, int lda,
                                  int m0, int k0, int lane) {
  int m  = m0 + (lane & 15);
  int kb = k0 + ((lane & 16) ? 8 : 0);
  const _Float16* row = A + (size_t)m * lda;
  v16h a;
#pragma unroll
  for (int i = 0; i < 8; ++i) a[i]     = row[kb + i];
#pragma unroll
  for (int i = 0; i < 8; ++i) a[8 + i] = row[kb + 16 + i];
  return a;
}

// Logical B[k][n] = W[n][k]  (W row-major [nrows x ldw]): B-tile from W rows.
__device__ inline v16h load_bT_f32(const float* __restrict__ W, int ldw,
                                   int n0, int k0, int lane) {
  int n  = n0 + (lane & 15);
  int kb = k0 + ((lane & 16) ? 8 : 0);
  const float* row = W + (size_t)n * ldw;
  v16h b;
#pragma unroll
  for (int i = 0; i < 8; ++i) b[i]     = (_Float16)row[kb + i];
#pragma unroll
  for (int i = 0; i < 8; ++i) b[8 + i] = (_Float16)row[kb + 16 + i];
  return b;
}

// ---------------------------------------------------------------------------
// Segment starts from sorted int64 batch ids. seg[g] = first node of graph g,
// seg[BG] = N. Handles empty graphs.
// ---------------------------------------------------------------------------
__global__ void seg_kernel(const long long* __restrict__ batch, int n_nodes,
                           int* __restrict__ seg) {
  int n = blockIdx.x * blockDim.x + threadIdx.x;
  if (n >= n_nodes) return;
  int bc = (int)batch[n];
  int bp = (n == 0) ? -1 : (int)batch[n - 1];
  for (int g = bp + 1; g <= bc; ++g) seg[g] = n;
  if (n == n_nodes - 1)
    for (int g = bc + 1; g <= BG; ++g) seg[g] = n_nodes;
}

// ---------------------------------------------------------------------------
// LSTM gate GEMM: gates[b][j] = bias_j + sum_k qstar[b][k] Wih[j][k]
//                              + sum_k h[b][k] Whh[j][k]
// 1024 tiles of 16x16 (64 M-tiles x 16 N-tiles), one wave each.
// ---------------------------------------------------------------------------
__global__ __launch_bounds__(128) void gates_kernel(
    const float* __restrict__ q_star, const float* __restrict__ h,
    const float* __restrict__ W_ih, const float* __restrict__ W_hh,
    const float* __restrict__ b_ih, const float* __restrict__ b_hh,
    float* __restrict__ gates) {
  int wave = threadIdx.x >> 5, lane = threadIdx.x & 31;
  int tile = blockIdx.x * 4 + wave;      // 0..1023
  int m0 = (tile >> 4) * 16;             // graph tile
  int n0 = (tile & 15) * 16;             // gate tile
  v8f c = {};
#pragma unroll
  for (int kk = 0; kk < 4; ++kk) {       // K = 128 (q_star)
    v16h a = load_a_f32(q_star, QS, m0, kk * 32, lane);
    v16h b = load_bT_f32(W_ih, QS, n0, kk * 32, lane);
    c = __builtin_amdgcn_wmma_f32_16x16x32_f16(false, a, false, b,
                                               (short)0, c, false, false);
  }
#pragma unroll
  for (int kk = 0; kk < 2; ++kk) {       // K = 64 (h)
    v16h a = load_a_f32(h, HDIM, m0, kk * 32, lane);
    v16h b = load_bT_f32(W_hh, HDIM, n0, kk * 32, lane);
    c = __builtin_amdgcn_wmma_f32_16x16x32_f16(false, a, false, b,
                                               (short)0, c, false, false);
  }
  int n = n0 + (lane & 15);
  float bias = b_ih[n] + b_hh[n];
  int mbase = m0 + ((lane & 16) ? 8 : 0);
#pragma unroll
  for (int v = 0; v < 8; ++v)
    gates[(size_t)(mbase + v) * GW + n] = c[v] + bias;
}

// ---------------------------------------------------------------------------
// Elementwise LSTM cell update; writes q (=h) into q_star[:, 0:64].
// ---------------------------------------------------------------------------
__global__ void lstm_update(const float* __restrict__ gates,
                            float* __restrict__ hbuf, float* __restrict__ cbuf,
                            float* __restrict__ q_star) {
  int idx = blockIdx.x * blockDim.x + threadIdx.x;   // BG*HDIM threads
  if (idx >= BG * HDIM) return;
  int b = idx >> 6, k = idx & 63;
  const float* gr = gates + (size_t)b * GW;
  float ig = 1.f / (1.f + __expf(-gr[k]));
  float fg = 1.f / (1.f + __expf(-gr[HDIM + k]));
  float gg = tanhf(gr[2 * HDIM + k]);
  float og = 1.f / (1.f + __expf(-gr[3 * HDIM + k]));
  float cn = fg * cbuf[idx] + ig * gg;
  float hn = og * tanhf(cn);
  cbuf[idx] = cn;
  hbuf[idx] = hn;
  q_star[(size_t)b * QS + k] = hn;                  // q half of q_star
}

// ---------------------------------------------------------------------------
// Attention: one block (8 waves) per graph, single streaming pass over the
// graph's contiguous node rows with online softmax.
// Lane l owns features 2l, 2l+1 (packed-friendly).
//   mode 0: read f32 x (non-temporal b64) and write packed-bf16 copy to x_h
//           (128 MB -> fits the 192 MB L2 for reuse by later steps)
//   mode 1: read packed-bf16 x_h (b32/lane, L2-resident)
//   mode 2: read f32 x only (fallback when ws can't hold x_h)
// ---------------------------------------------------------------------------
__global__ __launch_bounds__(256) void attn_kernel(
    const float* __restrict__ x, unsigned* __restrict__ xh,
    const int* __restrict__ seg, float* __restrict__ q_star, int mode) {
  int g = blockIdx.x;
  int s = seg[g], e = seg[g + 1];
  __shared__ float qsh[HDIM];
  __shared__ float wm[8], wsum[8], wr[8][HDIM];
  int tid = threadIdx.x;
  if (tid < HDIM) qsh[tid] = q_star[(size_t)g * QS + tid];
  __syncthreads();

  int wave = tid >> 5, lane = tid & 31;
  float q0 = qsh[2 * lane], q1 = qsh[2 * lane + 1];
  float m = -INFINITY, ssum = 0.f, r0 = 0.f, r1 = 0.f;

  if (mode == 1) {                       // L2-resident packed bf16 stream
    for (int n = s + wave; n < e; n += 8) {
      unsigned u = xh[(size_t)n * 32 + lane];
      if (n + 8 < e) __builtin_prefetch(xh + (size_t)(n + 8) * 32 + lane, 0, 1);
      float x0 = __uint_as_float(u << 16);
      float x1 = __uint_as_float(u & 0xffff0000u);
      float p = x0 * q0 + x1 * q1;
#pragma unroll
      for (int off = 16; off >= 1; off >>= 1) p += __shfl_xor(p, off, 32);
      float mnew  = fmaxf(m, p);
      float scale = __expf(m - mnew);
      float a     = __expf(p - mnew);
      ssum = ssum * scale + a;
      r0   = r0 * scale + a * x0;
      r1   = r1 * scale + a * x1;
      m = mnew;
    }
  } else {                               // f32 stream (NT), optional bf16 spill
    for (int n = s + wave; n < e; n += 8) {
      u64 d = __builtin_nontemporal_load(
                  (const u64*)(x + (size_t)n * HDIM + 2 * lane));
      float x0 = __uint_as_float((unsigned)d);
      float x1 = __uint_as_float((unsigned)(d >> 32));
      if (mode == 0) {                   // round-to-nearest bf16 pack
        unsigned b0 = (unsigned)d, b1 = (unsigned)(d >> 32);
        unsigned h0 = (b0 + 0x7fffu + ((b0 >> 16) & 1u)) >> 16;
        unsigned h1 = (b1 + 0x7fffu + ((b1 >> 16) & 1u)) >> 16;
        xh[(size_t)n * 32 + lane] = h0 | (h1 << 16);
      }
      float p = x0 * q0 + x1 * q1;
#pragma unroll
      for (int off = 16; off >= 1; off >>= 1) p += __shfl_xor(p, off, 32);
      float mnew  = fmaxf(m, p);
      float scale = __expf(m - mnew);
      float a     = __expf(p - mnew);
      ssum = ssum * scale + a;
      r0   = r0 * scale + a * x0;
      r1   = r1 * scale + a * x1;
      m = mnew;
    }
  }

  wr[wave][2 * lane]     = r0;
  wr[wave][2 * lane + 1] = r1;
  if (lane == 0) { wm[wave] = m; wsum[wave] = ssum; }
  __syncthreads();

  if (tid < HDIM) {
    float mstar = wm[0];
#pragma unroll
    for (int w = 1; w < 8; ++w) mstar = fmaxf(mstar, wm[w]);
    float stot = 0.f, rtot = 0.f;
#pragma unroll
    for (int w = 0; w < 8; ++w) {
      float sc = (wm[w] == -INFINITY) ? 0.f : __expf(wm[w] - mstar);
      stot += wsum[w] * sc;
      rtot += wr[w][tid] * sc;
    }
    q_star[(size_t)g * QS + HDIM + tid] = rtot / fmaxf(stot, 1e-16f);
  }
}

// ---------------------------------------------------------------------------
// MLP layer 1: h1 = relu(q_star @ W1^T + b1)  -> f16 [1024 x 256]
// ---------------------------------------------------------------------------
__global__ __launch_bounds__(128) void mlp1_kernel(
    const float* __restrict__ q_star, const float* __restrict__ W1,
    const float* __restrict__ b1, _Float16* __restrict__ h1) {
  int wave = threadIdx.x >> 5, lane = threadIdx.x & 31;
  int tile = blockIdx.x * 4 + wave;
  int m0 = (tile >> 4) * 16, n0 = (tile & 15) * 16;
  v8f c = {};
#pragma unroll
  for (int kk = 0; kk < 4; ++kk) {
    v16h a = load_a_f32(q_star, QS, m0, kk * 32, lane);
    v16h b = load_bT_f32(W1, QS, n0, kk * 32, lane);
    c = __builtin_amdgcn_wmma_f32_16x16x32_f16(false, a, false, b,
                                               (short)0, c, false, false);
  }
  int n = n0 + (lane & 15);
  float bias = b1[n];
  int mbase = m0 + ((lane & 16) ? 8 : 0);
#pragma unroll
  for (int v = 0; v < 8; ++v)
    h1[(size_t)(mbase + v) * 256 + n] = (_Float16)fmaxf(c[v] + bias, 0.f);
}

// ---------------------------------------------------------------------------
// MLP layer 2: out = h1 @ W2^T + b2  -> f32 [1024 x 128]
// ---------------------------------------------------------------------------
__global__ __launch_bounds__(128) void mlp2_kernel(
    const _Float16* __restrict__ h1, const float* __restrict__ W2,
    const float* __restrict__ b2, float* __restrict__ out) {
  int wave = threadIdx.x >> 5, lane = threadIdx.x & 31;
  int tile = blockIdx.x * 4 + wave;      // 0..511
  int m0 = (tile >> 3) * 16, n0 = (tile & 7) * 16;
  v8f c = {};
#pragma unroll
  for (int kk = 0; kk < 8; ++kk) {
    v16h a = load_a_f16(h1, 256, m0, kk * 32, lane);
    v16h b = load_bT_f32(W2, 256, n0, kk * 32, lane);
    c = __builtin_amdgcn_wmma_f32_16x16x32_f16(false, a, false, b,
                                               (short)0, c, false, false);
  }
  int n = n0 + (lane & 15);
  float bias = b2[n];
  int mbase = m0 + ((lane & 16) ? 8 : 0);
#pragma unroll
  for (int v = 0; v < 8; ++v)
    out[(size_t)(mbase + v) * 128 + n] = c[v] + bias;
}

// ---------------------------------------------------------------------------
extern "C" void kernel_launch(void* const* d_in, const int* in_sizes, int n_in,
                              void* d_out, int out_size, void* d_ws, size_t ws_size,
                              hipStream_t stream) {
  const float*     x     = (const float*)d_in[0];
  const long long* batch = (const long long*)d_in[1];
  const float*     W_ih  = (const float*)d_in[2];
  const float*     W_hh  = (const float*)d_in[3];
  const float*     b_ih  = (const float*)d_in[4];
  const float*     b_hh  = (const float*)d_in[5];
  const float*     W1    = (const float*)d_in[6];
  const float*     b1    = (const float*)d_in[7];
  const float*     W2    = (const float*)d_in[8];
  const float*     b2    = (const float*)d_in[9];
  float* out = (float*)d_out;

  const int n_nodes = in_sizes[0] / HDIM;

  // workspace layout (256B aligned slices)
  char* p = (char*)d_ws;
  auto take = [&](size_t bytes) { char* r = p; p += (bytes + 255) & ~(size_t)255; return r; };
  int*      seg    = (int*)take(sizeof(int) * (BG + 1));
  float*    q_star = (float*)take(sizeof(float) * BG * QS);
  float*    hbuf   = (float*)take(sizeof(float) * BG * HDIM);
  float*    cbuf   = (float*)take(sizeof(float) * BG * HDIM);
  float*    gates  = (float*)take(sizeof(float) * BG * GW);
  _Float16* h1     = (_Float16*)take(sizeof(_Float16) * BG * 256);
  size_t    base   = (size_t)(p - (char*)d_ws);
  size_t    xh_bytes = (size_t)n_nodes * HDIM * 2;   // packed bf16 copy, 128 MB
  bool      cache_ok = ws_size >= base + xh_bytes;
  unsigned* xh     = cache_ok ? (unsigned*)take(xh_bytes) : (unsigned*)q_star;

  // segment offsets + zero-init recurrent state (q_star,h,c are contiguous)
  seg_kernel<<<(n_nodes + 255) / 256, 256, 0, stream>>>(batch, n_nodes, seg);
  hipMemsetAsync(q_star, 0, sizeof(float) * BG * (QS + 2 * HDIM), stream);

  for (int step = 0; step < STEPS; ++step) {
    int mode = cache_ok ? (step == 0 ? 0 : 1) : 2;
    gates_kernel<<<256, 128, 0, stream>>>(q_star, hbuf, W_ih, W_hh, b_ih, b_hh, gates);
    lstm_update<<<(BG * HDIM) / 256, 256, 0, stream>>>(gates, hbuf, cbuf, q_star);
    attn_kernel<<<BG, 256, 0, stream>>>(x, xh, seg, q_star, mode);
  }

  mlp1_kernel<<<256, 128, 0, stream>>>(q_star, W1, b1, h1);
  mlp2_kernel<<<128, 128, 0, stream>>>(h1, W2, b2, out);
}
